// Head_40106404610133
// MI455X (gfx1250) — matrix operand order repaired and checked by reading, not compiled, not run on hardware
//
#include <hip/hip_runtime.h>
#include <hip/hip_bf16.h>
#include <math.h>

// Causal single-head attention for MI455X (gfx1250, wave32, WMMA).
// HBM-bound: x (201MB) read once by proj_kernel; q/k/v/P intermediates kept
// in bf16 and served from L2 (192MB) during the attention phase.
// proj_kernel pins its schedule with sched_barrier(0): all L2-hot B-fragment
// loads issue first, then the next K-step's HBM x loads, then the 12 WMMAs.
// In-order VMEM completion => B waits leave the x loads in flight, so HBM
// latency is hidden behind the WMMA block within a single wave.

#define BATCH 512
#define TLEN  256
#define CDIM  384
#define HDIM  64
#define BT    (BATCH * TLEN)

typedef __attribute__((ext_vector_type(16))) __bf16 v16bf;
typedef __attribute__((ext_vector_type(8)))  __bf16 v8bf;
typedef __attribute__((ext_vector_type(8)))  float  v8f;

union V16 { v16bf v; v8bf h[2]; };

__device__ __forceinline__ v8f wmma_bf16(v16bf a, v16bf b, v8f c) {
  // D = A(16x32 bf16) x B(32x16 bf16) + C(16x16 f32)
  return __builtin_amdgcn_wmma_f32_16x16x32_bf16(
      /*neg_a=*/false, a, /*neg_b=*/false, b,
      /*c_mod=*/(short)0, c, /*reuse_a=*/false, /*reuse_b=*/false);
}

// B-fragment loader: operand stored "N-major" (row n = output column, K
// contiguous). Lane holds N = lane%16, K = 16*(lane/16) + 0..15 -> one
// contiguous 32B chunk per lane (two b128 loads).
__device__ __forceinline__ v16bf loadB(const __bf16* base) {
  V16 B;
  const v8bf* p = reinterpret_cast<const v8bf*>(base);
  B.h[0] = p[0];
  B.h[1] = p[1];
  return B.v;
}

// Convert 16 raw f32 x-values (two 32B chunks) into an A-fragment.
__device__ __forceinline__ v16bf cvtA(const float4* u) {
  return (v16bf){(__bf16)u[0].x, (__bf16)u[0].y, (__bf16)u[0].z, (__bf16)u[0].w,
                 (__bf16)u[1].x, (__bf16)u[1].y, (__bf16)u[1].z, (__bf16)u[1].w,
                 (__bf16)u[2].x, (__bf16)u[2].y, (__bf16)u[2].z, (__bf16)u[2].w,
                 (__bf16)u[3].x, (__bf16)u[3].y, (__bf16)u[3].z, (__bf16)u[3].w};
}

// ---------------------------------------------------------------------------
// Kernel 0: W[C][H] f32 -> WT[3][H][C] bf16 (B-operand friendly layout)
// ---------------------------------------------------------------------------
__global__ void wt_kernel(const float* __restrict__ Wq,
                          const float* __restrict__ Wk,
                          const float* __restrict__ Wv,
                          __bf16* __restrict__ WT) {
  int id = blockIdx.x * blockDim.x + threadIdx.x;
  if (id >= 3 * HDIM * CDIM) return;
  int m    = id / (HDIM * CDIM);
  int rem  = id % (HDIM * CDIM);
  int hcol = rem / CDIM;
  int c    = rem % CDIM;
  const float* W = (m == 0) ? Wq : (m == 1) ? Wk : Wv;
  WT[id] = (__bf16)W[c * HDIM + hcol];
}

// ---------------------------------------------------------------------------
// Kernel 1: q/k/v projections. One wave = 16 rows of x, all three outputs
// accumulated from a single read of x (the dominant HBM traffic).
//   q_ws, k_ws : [B*T][H] bf16 row-major
//   vT_ws      : [B][H][T] bf16 (transposed for GEMM2 B-fragments)
// ---------------------------------------------------------------------------
__global__ void __launch_bounds__(256) proj_kernel(
    const float* __restrict__ x, const __bf16* __restrict__ WT,
    __bf16* __restrict__ q_ws, __bf16* __restrict__ k_ws,
    __bf16* __restrict__ vT_ws) {
  const int wave = threadIdx.x >> 5;
  const int lane = threadIdx.x & 31;
  const int lid  = lane & 15;   // M (A/C row group) or N
  const int hh   = lane >> 4;   // lane half
  const int row0 = (blockIdx.x * 8 + wave) * 16;

  const float* xrow = x + (size_t)(row0 + lid) * CDIM;
  const __bf16* WTq = WT;
  const __bf16* WTk = WT + HDIM * CDIM;
  const __bf16* WTv = WT + 2 * HDIM * CDIM;

  v8f aq[4], ak[4], av[4];
#pragma unroll
  for (int n = 0; n < 4; ++n) { aq[n] = {}; ak[n] = {}; av[n] = {}; }

  // Prime the pipeline: raw x loads for kc=0.
  float4 araw[4];
  {
    const float4* p0 = reinterpret_cast<const float4*>(xrow + 16 * hh);
    const float4* p1 = reinterpret_cast<const float4*>(xrow + 16 * hh + 8);
    araw[0] = p0[0]; araw[1] = p0[1]; araw[2] = p1[0]; araw[3] = p1[1];
  }
  v16bf A = cvtA(araw);

  for (int kc = 0; kc < CDIM / 32; ++kc) {
    const int koff = kc * 32 + 16 * hh;
    // (1) Issue ALL B-fragment loads for this K-step (L1/L2-hot).
    v16bf Bq[4], Bk[4], Bv[4];
#pragma unroll
    for (int n = 0; n < 4; ++n) {
      const size_t off = (size_t)(n * 16 + lid) * CDIM + koff;
      Bq[n] = loadB(WTq + off);
      Bk[n] = loadB(WTk + off);
      Bv[n] = loadB(WTv + off);
    }
    // Pin: B loads may not sink below this point.
    __builtin_amdgcn_sched_barrier(0);
    // (2) Issue next K-step's HBM x loads AFTER the B loads: in-order VMEM
    //     completion => no B wait ever forces these to complete. Clamped
    //     index keeps this a single basic block (redundant reload on last
    //     iteration, result unused).
    const int kcn = (kc + 1 < CDIM / 32) ? kc + 1 : kc;
    float4 anext[4];
    {
      const float4* p0 =
          reinterpret_cast<const float4*>(xrow + kcn * 32 + 16 * hh);
      const float4* p1 =
          reinterpret_cast<const float4*>(xrow + kcn * 32 + 16 * hh + 8);
      anext[0] = p0[0]; anext[1] = p0[1]; anext[2] = p1[0]; anext[3] = p1[1];
    }
    __builtin_amdgcn_sched_barrier(0);
    // (3) 12 WMMAs: first waits only on B (x loads stay in flight), the
    //     rest issue back-to-back from registers.
#pragma unroll
    for (int n = 0; n < 4; ++n) {
      aq[n] = wmma_bf16(A, Bq[n], aq[n]);
      ak[n] = wmma_bf16(A, Bk[n], ak[n]);
      av[n] = wmma_bf16(A, Bv[n], av[n]);
    }
    A = cvtA(anext);  // waits for the prefetch only here
  }

  // D layout: element r of lane -> (M = hh*8 + r, N = lid)
#pragma unroll
  for (int n = 0; n < 4; ++n) {
#pragma unroll
    for (int r = 0; r < 8; ++r) {
      const int row = row0 + hh * 8 + r;
      const int col = n * 16 + lid;
      q_ws[(size_t)row * HDIM + col] = (__bf16)aq[n][r];
      k_ws[(size_t)row * HDIM + col] = (__bf16)ak[n][r];
      const int bb = row >> 8;   // / TLEN
      const int tt = row & 255;  // % TLEN
      vT_ws[((size_t)bb * HDIM + col) * TLEN + tt] = (__bf16)av[n][r];
    }
  }
}

// ---------------------------------------------------------------------------
// Kernel 2: attention. Block = 4 waves; wave owns 16 query rows of one batch.
// Full 256-wide score row in 16 f32 accumulators; softmax via 16-lane
// shfl_xor butterflies; P restaged through padded LDS (D-layout -> A-layout).
// ---------------------------------------------------------------------------
__global__ void __launch_bounds__(128) attn_kernel(
    const __bf16* __restrict__ q_ws, const __bf16* __restrict__ k_ws,
    const __bf16* __restrict__ vT_ws, float* __restrict__ out) {
  __shared__ __bf16 ps[4][16][40];  // per-wave 16x32 P tile, rows padded to 40
                                    // (80B stride -> conflict-free b128 reads)
  const int wave = threadIdx.x >> 5;
  const int lane = threadIdx.x & 31;
  const int lid  = lane & 15;
  const int hh   = lane >> 4;
  const int b    = blockIdx.x >> 2;
  const int tb   = blockIdx.x & 3;
  const int t0   = tb * 64 + wave * 16;
  const int nmax = t0 >> 4;  // last s-tile touching the causal triangle
  const float scale = 0.05103103630798288f;  // 384^-0.5 (matches reference)

  // q A-fragments for K(head)=0..31 and 32..63
  V16 qa0, qa1;
  {
    const __bf16* qrow = q_ws + (size_t)(b * TLEN + t0 + lid) * HDIM;
    qa0.h[0] = *reinterpret_cast<const v8bf*>(qrow + 16 * hh);
    qa0.h[1] = *reinterpret_cast<const v8bf*>(qrow + 8 + 16 * hh);
    qa1.h[0] = *reinterpret_cast<const v8bf*>(qrow + 32 + 16 * hh);
    qa1.h[1] = *reinterpret_cast<const v8bf*>(qrow + 40 + 16 * hh);
  }

  v8f acc[16];
#pragma unroll
  for (int n = 0; n < 16; ++n) acc[n] = {};

  // GEMM1: wei = q @ k^T ; B-fragments straight from row-major k (L2-hot).
#pragma unroll
  for (int n = 0; n < 16; ++n) {
    if (n <= nmax) {  // wave-uniform guard: EXEC stays all-1s around WMMA
      const __bf16* krow = k_ws + (size_t)(b * TLEN + n * 16 + lid) * HDIM;
      acc[n] = wmma_bf16(qa0.v, loadB(krow + 16 * hh), acc[n]);
      acc[n] = wmma_bf16(qa1.v, loadB(krow + 32 + 16 * hh), acc[n]);
    }
  }

  // Scale + causal mask (D element r -> t = t0 + 8*hh + r, s = n*16 + lid)
#pragma unroll
  for (int n = 0; n < 16; ++n) {
#pragma unroll
    for (int r = 0; r < 8; ++r) {
      const int t = t0 + hh * 8 + r;
      const int s = n * 16 + lid;
      const float w = acc[n][r] * scale;
      acc[n][r] = (s > t) ? -INFINITY : w;
    }
  }

  // Softmax: row lives in 16 lanes of one half x 16 tiles at fixed r.
  float rinv[8];
#pragma unroll
  for (int r = 0; r < 8; ++r) {
    float m = -INFINITY;
#pragma unroll
    for (int n = 0; n < 16; ++n)
      if (n <= nmax) m = fmaxf(m, acc[n][r]);
    m = fmaxf(m, __shfl_xor(m, 1, 32));
    m = fmaxf(m, __shfl_xor(m, 2, 32));
    m = fmaxf(m, __shfl_xor(m, 4, 32));
    m = fmaxf(m, __shfl_xor(m, 8, 32));
    float sum = 0.f;
#pragma unroll
    for (int n = 0; n < 16; ++n) {
      if (n <= nmax) {
        const float p = __expf(acc[n][r] - m);  // exp(-inf - m) = 0 masks
        acc[n][r] = p;
        sum += p;
      }
    }
    sum += __shfl_xor(sum, 1, 32);
    sum += __shfl_xor(sum, 2, 32);
    sum += __shfl_xor(sum, 4, 32);
    sum += __shfl_xor(sum, 8, 32);
    rinv[r] = 1.0f / sum;  // 8 reciprocals instead of 32 divisions
  }

  // GEMM2: out = P @ v, K(s) in chunks of 32 (two score tiles per chunk).
  v8f o[4];
#pragma unroll
  for (int n2 = 0; n2 < 4; ++n2) o[n2] = {};

  const int nchunks = (nmax + 2) >> 1;
#pragma unroll
  for (int kc = 0; kc < 8; ++kc) {
    if (kc < nchunks) {  // wave-uniform
      // Stage P chunk [16 x 32] bf16 into per-wave LDS (D-layout writes).
#pragma unroll
      for (int j = 0; j < 2; ++j) {
        const int n = kc * 2 + j;
#pragma unroll
        for (int r = 0; r < 8; ++r) {
          const float p = (n <= nmax) ? acc[n][r] : 0.f;
          ps[wave][hh * 8 + r][j * 16 + lid] = (__bf16)p;
        }
      }
      // CDNA5 split counter: all LDS stores visible before A-frag reads.
      asm volatile("s_wait_dscnt 0x0" ::: "memory");
      V16 pa;
      {
        const __bf16* prow = &ps[wave][lid][0];
        pa.h[0] = *reinterpret_cast<const v8bf*>(prow + 16 * hh);
        pa.h[1] = *reinterpret_cast<const v8bf*>(prow + 8 + 16 * hh);
      }
#pragma unroll
      for (int n2 = 0; n2 < 4; ++n2) {
        const __bf16* vrow =
            vT_ws + (size_t)(b * HDIM + n2 * 16 + lid) * TLEN + kc * 32 + 16 * hh;
        o[n2] = wmma_bf16(pa.v, loadB(vrow), o[n2]);
      }
    }
  }

  // Normalize by softmax row sum and store f32 output (coalesced 64B rows).
#pragma unroll
  for (int n2 = 0; n2 < 4; ++n2) {
#pragma unroll
    for (int r = 0; r < 8; ++r) {
      const int t    = t0 + hh * 8 + r;
      const int head = n2 * 16 + lid;
      out[((size_t)(b * TLEN + t)) * HDIM + head] = o[n2][r] * rinv[r];
    }
  }
}

// ---------------------------------------------------------------------------
extern "C" void kernel_launch(void* const* d_in, const int* in_sizes, int n_in,
                              void* d_out, int out_size, void* d_ws,
                              size_t ws_size, hipStream_t stream) {
  const float* x  = (const float*)d_in[0];
  const float* Wk = (const float*)d_in[1];
  const float* Wq = (const float*)d_in[2];
  const float* Wv = (const float*)d_in[3];
  float* out = (float*)d_out;

  // Workspace: q | k | vT (each B*T*H bf16) | WT (3*H*C bf16)  ~= 50.5 MB
  __bf16* q_ws  = (__bf16*)d_ws;
  __bf16* k_ws  = q_ws + (size_t)BT * HDIM;
  __bf16* vT_ws = k_ws + (size_t)BT * HDIM;
  __bf16* WT    = vT_ws + (size_t)BT * HDIM;

  wt_kernel<<<(3 * HDIM * CDIM + 255) / 256, 256, 0, stream>>>(Wq, Wk, Wv, WT);
  proj_kernel<<<BT / 16 / 8, 256, 0, stream>>>(x, WT, q_ws, k_ws, vT_ws);
  attn_kernel<<<BATCH * 4, 128, 0, stream>>>(q_ws, k_ws, vT_ws, out);
}
